// GraphNetFeature_7121055777569
// MI455X (gfx1250) — compile-verified, matrix operand order checked by reading, not run on hardware
//
#include <hip/hip_runtime.h>
#include <hip/hip_bf16.h>

typedef __attribute__((ext_vector_type(16))) __bf16 bf16x16;
typedef __attribute__((ext_vector_type(8)))  float  floatx8;

#define BN_EPS 1e-5f
#define NEG_BIG (-3.0e38f)

// Low 32 bits of a generic pointer to LDS == LDS byte offset (ISA 10.2 aperture map)
__device__ __forceinline__ unsigned lds_off(const void* p) {
    return (unsigned)(unsigned long long)p;
}

// One lane: async-copy 16 bytes global -> LDS, tracked by ASYNCcnt.
__device__ __forceinline__ void async_g2l_b128(unsigned lds, unsigned long long gaddr) {
    asm volatile("global_load_async_to_lds_b128 %0, %1, off"
                 :: "v"(lds), "v"(gaddr) : "memory");
}
__device__ __forceinline__ void wait_asynccnt0() {
    asm volatile("s_wait_asynccnt 0x0" ::: "memory");
}

// ---------------------------------------------------------------------------
// Generic elementwise f32 -> bf16 convert (weights)
// ---------------------------------------------------------------------------
__global__ void k_f32_to_bf16(const float* __restrict__ s, __bf16* __restrict__ d, long n) {
    long i = (long)blockIdx.x * blockDim.x + threadIdx.x;
    long stride = (long)gridDim.x * blockDim.x;
    for (; i < n; i += stride) d[i] = (__bf16)s[i];
}

// ---------------------------------------------------------------------------
// x (B, C, N) f32  ->  xb (B*N, C) bf16   (transpose + convert)
// ---------------------------------------------------------------------------
__global__ void k_transpose_bf16(const float* __restrict__ x, __bf16* __restrict__ dst,
                                 int C, int N, long total) {
    long i = (long)blockIdx.x * blockDim.x + threadIdx.x;
    if (i >= total) return;
    int n = (int)(i % N);
    int c = (int)((i / N) % C);
    int b = (int)(i / ((long)N * C));
    dst[((long)b * N + n) * C + c] = (__bf16)x[i];
}

// ---------------------------------------------------------------------------
// Row squared-norms from bf16 rows (one wave32 per row)
// ---------------------------------------------------------------------------
__global__ __launch_bounds__(32)
void k_rownorm_bf16(const __bf16* __restrict__ xb, int C, float* __restrict__ xx) {
    long r = blockIdx.x;
    const __bf16* p = xb + r * (long)C;
    float acc = 0.f;
    for (int c = threadIdx.x; c < C; c += 32) { float v = (float)p[c]; acc += v * v; }
    for (int off = 16; off > 0; off >>= 1) acc += __shfl_xor(acc, off, 32);
    if (threadIdx.x == 0) xx[r] = acc;
}

// ---------------------------------------------------------------------------
// f32 slice (row stride lds) -> bf16 rows (stride C) + row norms
// ---------------------------------------------------------------------------
__global__ __launch_bounds__(32)
void k_convert_norm(const float* __restrict__ src, int ldsrc,
                    __bf16* __restrict__ dst, int C, float* __restrict__ xx) {
    long r = blockIdx.x;
    const float* s = src + r * (long)ldsrc;
    __bf16* d = dst + r * (long)C;
    float acc = 0.f;
    for (int c = threadIdx.x; c < C; c += 32) {
        __bf16 bv = (__bf16)s[c];
        d[c] = bv;
        float f = (float)bv;
        acc += f * f;
    }
    for (int off = 16; off > 0; off >>= 1) acc += __shfl_xor(acc, off, 32);
    if (threadIdx.x == 0) xx[r] = acc;
}

// ---------------------------------------------------------------------------
// Tiled NT GEMM, bf16 operands, fp32 accumulate via v_wmma_f32_16x16x32_bf16.
//   C[m,n] = sum_k A[m,k] * B[n,k]        (A, B row-major bf16)
// 128 threads = 4 waves; block tile 64(M) x 64(N).
// K staged 64x32 per buffer with DOUBLE-BUFFERED global_load_async_to_lds_b128:
// tile t+1's async copies are issued right after the barrier, overlapping the
// ds_load + v_wmma work on tile t; ASYNCcnt gates buffer reuse.
// MODE 0: plain store   MODE 1: negdist (2S - xx[m] - xx[n])   MODE 2: BN+LReLU
// ---------------------------------------------------------------------------
enum { GM_STORE = 0, GM_NEGDIST = 1, GM_BNLRELU = 2 };

template <int MODE>
__global__ __launch_bounds__(128)
void gemm_nt_bf16(const __bf16* __restrict__ A, const __bf16* __restrict__ Bm,
                  float* __restrict__ C,
                  int K, int lda, int ldb, int ldc,
                  long strideA, long strideB, long strideC,
                  const float* __restrict__ xx, long strideXX,
                  const float* __restrict__ bn, int bnC) {
    __shared__ __attribute__((aligned(32))) __bf16 lA[2][64 * 32];
    __shared__ __attribute__((aligned(32))) __bf16 lB[2][64 * 32];

    const int bz = blockIdx.z;
    A  += (long)bz * strideA;
    Bm += (long)bz * strideB;
    C  += (long)bz * strideC;
    const float* xxb = (MODE == GM_NEGDIST) ? (xx + (long)bz * strideXX) : nullptr;

    const int tid  = threadIdx.x;
    const int wave = tid >> 5;
    const int lane = tid & 31;
    const int half = lane >> 4;   // which 16-wide K half this lane carries
    const int l16  = lane & 15;
    const int row0 = blockIdx.x * 64;
    const int col0 = blockIdx.y * 64;

    // Staging assignment: 128 threads * 32B each = one 64x32 bf16 tile
    const int srow = tid >> 1;
    const int scol = (tid & 1) * 16;                 // element offset (16 elem = 32B)

    const __bf16* gA = A  + (long)(row0 + srow) * lda + scol;
    const __bf16* gB = Bm + (long)(col0 + srow) * ldb + scol;
    const unsigned dA0 = lds_off(&lA[0][srow * 32 + scol]);
    const unsigned dA1 = lds_off(&lA[1][srow * 32 + scol]);
    const unsigned dB0 = lds_off(&lB[0][srow * 32 + scol]);
    const unsigned dB1 = lds_off(&lB[1][srow * 32 + scol]);

    floatx8 acc[4] = {};

    // issue this thread's 4 x 16B async copies for K-tile at k0 into buffer `buf`
    auto issue = [&](int k0, int buf) {
        unsigned long long a = (unsigned long long)(gA + k0);
        unsigned long long b = (unsigned long long)(gB + k0);
        unsigned da = buf ? dA1 : dA0;
        unsigned db = buf ? dB1 : dB0;
        async_g2l_b128(da,       a);
        async_g2l_b128(da + 16u, a + 16ull);
        async_g2l_b128(db,       b);
        async_g2l_b128(db + 16u, b + 16ull);
    };

    issue(0, 0);  // prologue: stage tile 0

    const int kTiles = K >> 5;
    for (int t = 0; t < kTiles; ++t) {
        const int cur = t & 1;
        wait_asynccnt0();   // my copies for tile t have landed in LDS
        __syncthreads();    // everyone's copies landed; previous readers done

        if (t + 1 < kTiles) {
            issue((t + 1) << 5, cur ^ 1);                 // overlap with compute
            if (t + 2 < kTiles) {                         // warm GL2 one more ahead
                __builtin_prefetch(gA + ((t + 2) << 5), 0, 0);
                __builtin_prefetch(gB + ((t + 2) << 5), 0, 0);
            }
        }

        // A fragment: lane carries row (wave*16 + l16), K-half selected by lane/16
        bf16x16 afrag = *(const bf16x16*)(&lA[cur][(wave * 16 + l16) * 32 + half * 16]);
#pragma unroll
        for (int j = 0; j < 4; ++j) {
            bf16x16 bfrag = *(const bf16x16*)(&lB[cur][(j * 16 + l16) * 32 + half * 16]);
            acc[j] = __builtin_amdgcn_wmma_f32_16x16x32_bf16(
                false, afrag, false, bfrag, (short)0, acc[j], false, false);
        }
    }

    // D layout: reg r of lane l -> M = r + 8*(l/16), N = l%16
#pragma unroll
    for (int j = 0; j < 4; ++j) {
#pragma unroll
        for (int r = 0; r < 8; ++r) {
            int row = row0 + wave * 16 + half * 8 + r;
            int col = col0 + j * 16 + l16;
            float v = acc[j][r];
            if (MODE == GM_NEGDIST) {
                v = 2.0f * v - xxb[row] - xxb[col];
            } else if (MODE == GM_BNLRELU) {
                float g = bn[col], bb = bn[bnC + col], m = bn[2 * bnC + col], vv = bn[3 * bnC + col];
                float s = g * rsqrtf(vv + BN_EPS);
                v = (v - m) * s + bb;
                v = v > 0.f ? v : 0.2f * v;
            }
            C[(long)row * ldc + col] = v;
        }
    }
}

// ---------------------------------------------------------------------------
// Top-20 (argmax-by-rounds) per row of the 1024-wide neg-dist matrix.
// One wave32 per row; 32 values per lane held in registers; butterfly argmax.
// ---------------------------------------------------------------------------
__global__ __launch_bounds__(32)
void k_topk20(const float* __restrict__ nd, int* __restrict__ idx) {
    long row = blockIdx.x;
    const float* d = nd + row * 1024;
    const int lane = threadIdx.x;

    float vals[32];
#pragma unroll
    for (int i = 0; i < 32; ++i) vals[i] = d[lane + 32 * i];

    unsigned taken = 0u;
    for (int t = 0; t < 20; ++t) {
        float best = NEG_BIG;
        int bi = 0;
#pragma unroll
        for (int i = 0; i < 32; ++i) {
            bool ok = !((taken >> i) & 1u);
            float v = vals[i];
            if (ok && v > best) { best = v; bi = i; }
        }
        float bv = best;
        int bg = bi * 32 + lane;  // global element index
        for (int off = 16; off > 0; off >>= 1) {
            float ov = __shfl_xor(bv, off, 32);
            int   og = __shfl_xor(bg, off, 32);
            if (ov > bv || (ov == bv && og < bg)) { bv = ov; bg = og; }
        }
        if ((bg & 31) == lane) taken |= 1u << (bg >> 5);
        if (lane == 0) idx[row * 20 + t] = bg;
    }
}

// ---------------------------------------------------------------------------
// Fused gather-max over 20 neighbors + affine recombination + BN + LReLU.
// out[n,c] = lrelu( s_c*( max_k G[idx_k,c] - G[n,c] + H[n,c] - m_c ) + b_c )
// ---------------------------------------------------------------------------
__global__ __launch_bounds__(256)
void k_gather_max_bn(const float* __restrict__ G, const float* __restrict__ H,
                     const int* __restrict__ idx, const float* __restrict__ bn,
                     float* __restrict__ outBase, int O, int ldo, int colOff, int N) {
    int p = blockIdx.x;            // 0 .. B*N-1
    int b = p / N;
    __shared__ int sidx[20];
    if (threadIdx.x < 20) sidx[threadIdx.x] = idx[(long)p * 20 + threadIdx.x];
    __syncthreads();

    for (int c = threadIdx.x; c < O; c += 256) {
        float mx = NEG_BIG;
#pragma unroll 4
        for (int k = 0; k < 20; ++k)
            mx = fmaxf(mx, G[((long)b * N + sidx[k]) * O + c]);
        float y = mx - G[(long)p * O + c] + H[(long)p * O + c];
        float g = bn[c], bb = bn[O + c], m = bn[2 * O + c], vv = bn[3 * O + c];
        float s = g * rsqrtf(vv + BN_EPS);
        y = (y - m) * s + bb;
        y = y > 0.f ? y : 0.2f * y;
        outBase[(long)p * ldo + colOff + c] = y;
    }
}

// ---------------------------------------------------------------------------
// Global max + mean pool over the N axis of h5 (B, N, 1024) -> pooled (B, 2048)
// ---------------------------------------------------------------------------
__global__ void k_pool(const float* __restrict__ h5, float* __restrict__ pooled,
                       int N, int C) {
    int c = blockIdx.x * blockDim.x + threadIdx.x;
    int b = blockIdx.y;
    if (c >= C) return;
    const float* p = h5 + (long)b * N * C + c;
    float mx = NEG_BIG, sm = 0.f;
    for (int n = 0; n < N; ++n) {
        float v = p[(long)n * C];
        mx = fmaxf(mx, v);
        sm += v;
    }
    pooled[(long)b * 2 * C + c]     = mx;
    pooled[(long)b * 2 * C + C + c] = sm / (float)N;
}

// ---------------------------------------------------------------------------
// Small FC: z = lrelu(bn6(pooled @ Wl1^T))    (4 x 2048) -> (4 x 512)
// ---------------------------------------------------------------------------
__global__ void k_fc_bn_lrelu(const float* __restrict__ pooled, const float* __restrict__ W,
                              const float* __restrict__ bn, float* __restrict__ z,
                              int In, int Out, int Btot) {
    int o = blockIdx.x * blockDim.x + threadIdx.x;
    if (o >= Btot * Out) return;
    int b = o / Out, oo = o % Out;
    const float* p = pooled + (long)b * In;
    const float* w = W + (long)oo * In;
    float acc = 0.f;
    for (int i = 0; i < In; ++i) acc += p[i] * w[i];
    float g = bn[oo], bb = bn[Out + oo], m = bn[2 * Out + oo], vv = bn[3 * Out + oo];
    float s = g * rsqrtf(vv + BN_EPS);
    float y = (acc - m) * s + bb;
    z[(long)b * Out + oo] = y > 0.f ? y : 0.2f * y;
}

// out = z @ Wl3^T + bl3     (4 x 512) -> (4 x 40)
__global__ void k_fc_bias(const float* __restrict__ z, const float* __restrict__ W,
                          const float* __restrict__ bias, float* __restrict__ out,
                          int In, int Out, int Btot) {
    int o = blockIdx.x * blockDim.x + threadIdx.x;
    if (o >= Btot * Out) return;
    int b = o / Out, oo = o % Out;
    const float* p = z + (long)b * In;
    const float* w = W + (long)oo * In;
    float acc = 0.f;
    for (int i = 0; i < In; ++i) acc += p[i] * w[i];
    out[(long)b * Out + oo] = acc + bias[oo];
}

// ---------------------------------------------------------------------------
// Host-side launch
// ---------------------------------------------------------------------------
extern "C" void kernel_launch(void* const* d_in, const int* in_sizes, int n_in,
                              void* d_out, int out_size, void* d_ws, size_t ws_size,
                              hipStream_t stream) {
    const int B = 4, N = 1024;
    const long BN = (long)B * N;

    // ---- workspace carve-out (all offsets 256B aligned) ----
    char* ws = (char*)d_ws;
    size_t off = 0;
    auto take = [&](size_t bytes) -> char* {
        char* p = ws + off;
        off = (off + bytes + 255) & ~(size_t)255;
        return p;
    };
    float*  nd    = (float*)take((size_t)B * N * N * 4);     // neg-dist (B,N,N)
    int*    idxb  = (int*)  take((size_t)BN * 20 * 4);       // knn indices
    __bf16* xb    = (__bf16*)take((size_t)BN * 1152 * 2);    // bf16 layer input / h
    float*  xx    = (float*)take((size_t)BN * 4);            // row norms
    float*  Gb    = (float*)take((size_t)BN * 512 * 4);      // G = X Wa^T
    float*  Hb    = (float*)take((size_t)BN * 512 * 4);      // H = X Wb^T
    float*  hbuf  = (float*)take((size_t)BN * 1152 * 4);     // concat(x1..x4)
    float*  h5    = (float*)take((size_t)BN * 1024 * 4);     // after W5+bn5
    float*  pooled= (float*)take((size_t)B * 2048 * 4);
    float*  zbuf  = (float*)take((size_t)B * 512 * 4);
    __bf16* wb1   = (__bf16*)take((size_t)512 * 1536 * 2);
    __bf16* wb2   = (__bf16*)take((size_t)256 * 1024 * 2);
    __bf16* wb3   = (__bf16*)take((size_t)128 * 512 * 2);
    __bf16* wb4   = (__bf16*)take((size_t)256 * 256 * 2);
    __bf16* wb5   = (__bf16*)take((size_t)1024 * 1152 * 2);

    const float* x   = (const float*)d_in[0];
    const float* Wf[5]  = {(const float*)d_in[1], (const float*)d_in[3],
                           (const float*)d_in[5], (const float*)d_in[7],
                           (const float*)d_in[9]};
    const float* bnp[5] = {(const float*)d_in[2], (const float*)d_in[4],
                           (const float*)d_in[6], (const float*)d_in[8],
                           (const float*)d_in[10]};
    const float* Wl1 = (const float*)d_in[11];
    const float* bn6 = (const float*)d_in[12];
    const float* Wl3 = (const float*)d_in[13];
    const float* bl3 = (const float*)d_in[14];
    float* outp = (float*)d_out;

    // ---- convert weights to bf16 ----
    __bf16* wb[5] = {wb1, wb2, wb3, wb4, wb5};
    long wsz[5] = {512L * 1536, 256L * 1024, 128L * 512, 256L * 256, 1024L * 1152};
    for (int i = 0; i < 5; ++i)
        k_f32_to_bf16<<<1024, 256, 0, stream>>>(Wf[i], wb[i], wsz[i]);

    // ---- layer 1 input: transpose x (B,C,N) -> xb (B*N, 768) bf16, row norms ----
    {
        long total = (long)B * 768 * N;
        k_transpose_bf16<<<(int)((total + 255) / 256), 256, 0, stream>>>(x, xb, 768, N, total);
        k_rownorm_bf16<<<(int)BN, 32, 0, stream>>>(xb, 768, xx);
    }

    const int Cdim[4]   = {768, 512, 256, 128};
    const int Odim[4]   = {512, 256, 128, 256};
    const int colOff[4] = {0, 512, 768, 896};

    for (int L = 0; L < 4; ++L) {
        const int C = Cdim[L], O = Odim[L];

        // (a) neg-dist Gram: nd = 2*X X^T - xx_m - xx_n  (per batch, WMMA bf16)
        gemm_nt_bf16<GM_NEGDIST><<<dim3(16, 16, 4), 128, 0, stream>>>(
            xb, xb, nd, C, C, C, N,
            (long)N * C, (long)N * C, (long)N * N,
            xx, (long)N, nullptr, 0);

        // (b) top-20 neighbor indices per point
        k_topk20<<<(int)BN, 32, 0, stream>>>(nd, idxb);

        // (c) G = X Wa^T, H = X Wb^T  (Wa/Wb = left/right halves of W rows)
        gemm_nt_bf16<GM_STORE><<<dim3(64, O / 64, 1), 128, 0, stream>>>(
            xb, wb[L], Gb, C, C, 2 * C, O, 0, 0, 0, nullptr, 0, nullptr, 0);
        gemm_nt_bf16<GM_STORE><<<dim3(64, O / 64, 1), 128, 0, stream>>>(
            xb, wb[L] + C, Hb, C, C, 2 * C, O, 0, 0, 0, nullptr, 0, nullptr, 0);

        // (d) gather-max + BN + LReLU -> slice of hbuf
        k_gather_max_bn<<<(int)BN, 256, 0, stream>>>(
            Gb, Hb, idxb, bnp[L], hbuf, O, 1152, colOff[L], N);

        // (e) prep next layer's bf16 input + norms
        if (L < 3)
            k_convert_norm<<<(int)BN, 32, 0, stream>>>(
                hbuf + colOff[L], 1152, xb, O, xx);
    }

    // ---- h (B*N,1152) -> bf16, W5 GEMM with fused bn5+lrelu -> h5 (B*N,1024) ----
    k_convert_norm<<<(int)BN, 32, 0, stream>>>(hbuf, 1152, xb, 1152, xx);
    gemm_nt_bf16<GM_BNLRELU><<<dim3(64, 16, 1), 128, 0, stream>>>(
        xb, wb5, h5, 1152, 1152, 1152, 1024, 0, 0, 0, nullptr, 0, bnp[4], 1024);

    // ---- global max+mean pool, FC head ----
    k_pool<<<dim3(4, 4), 256, 0, stream>>>(h5, pooled, N, 1024);
    k_fc_bn_lrelu<<<(B * 512 + 255) / 256, 256, 0, stream>>>(
        pooled, Wl1, bn6, zbuf, 2048, 512, B);
    k_fc_bias<<<(B * 40 + 255) / 256, 256, 0, stream>>>(
        zbuf, Wl3, bl3, outp, 512, 40, B);

    (void)in_sizes; (void)n_in; (void)out_size; (void)ws_size;
}
